// DeepSeekMoE_40956808134763
// MI455X (gfx1250) — matrix operand order repaired and checked by reading, not compiled, not run on hardware
//
#include <hip/hip_runtime.h>
#include <hip/hip_bf16.h>

// Problem constants (match reference)
static constexpr int E = 8;
static constexpr int D = 2048;
static constexpr int H = 1408;
static constexpr int T = 4096;        // B*S = 2*2048
static constexpr int HPAD = 1416;     // LDS row pad: 708 dwords % 64 banks = 4
static constexpr int NWAVES = 4;      // 128 threads / wave32
static constexpr int MT = 32;         // tokens per block (2 M-subtiles of 16)

typedef __attribute__((ext_vector_type(16))) __bf16 v16bf;
typedef __attribute__((ext_vector_type(8)))  float  v8f;

// ---------- helpers ----------
__device__ __forceinline__ unsigned short f2bf(float f) {
    union { float f; unsigned int u; } v; v.f = f;
    unsigned int u = v.u;
    u += 0x7FFFu + ((u >> 16) & 1u);   // round-to-nearest-even
    return (unsigned short)(u >> 16);
}

// A fragment: 16x32 bf16.  lane: m = lane&15, half = lane>>4.
// VGPR0-3 : K = half*8 + 0..7      VGPR4-7 : K = 16 + half*8 + 0..7
__device__ __forceinline__ v16bf load_fragA(const unsigned short* rowPtr, int k0, int half) {
    const unsigned short* p = rowPtr + k0 + half * 8;
    union { v16bf v; uint4 q[2]; } u;
    u.q[0] = *(const uint4*)(p);
    u.q[1] = *(const uint4*)(p + 16);
    return u.v;
}

// B fragment: 32x16 bf16. lane: n = lane&15, half = lane>>4 holds K = half*8 + 0..7, 16 + half*8 + 0..7
__device__ __forceinline__ v16bf load_fragB(const unsigned short* rowPtr, int k0, int half) {
    const unsigned short* p = rowPtr + k0 + half * 8;
    union { v16bf v; uint4 q[2]; } u;
    u.q[0] = *(const uint4*)(p);
    u.q[1] = *(const uint4*)(p + 16);
    return u.v;
}

__device__ __forceinline__ v8f wmma_bf16(v16bf a, v16bf b, v8f c) {
    return __builtin_amdgcn_wmma_f32_16x16x32_bf16(false, a, false, b,
                                                   (short)0, c, false, false);
}

// 2 M-subtiles x NS N-subtiles register-tiled GEMM inner loop.
// A rows: aR0 (rows 0..15), aR1 (rows 16..31); B rows start at bBase with
// bStride elements between consecutive 16-row groups; K extent KK.
template <int NS>
__device__ __forceinline__ void gemm2xN(const unsigned short* aR0,
                                        const unsigned short* aR1,
                                        const unsigned short* bBase, size_t bStride,
                                        int KK, int half, v8f (&c)[2][NS]) {
    const unsigned short* bR[NS];
#pragma unroll
    for (int j = 0; j < NS; ++j) bR[j] = bBase + (size_t)j * bStride;
    for (int k0 = 0; k0 < KK; k0 += 32) {
        v16bf a0 = load_fragA(aR0, k0, half);
        v16bf a1 = load_fragA(aR1, k0, half);
#pragma unroll
        for (int j = 0; j < NS; ++j) {
            v16bf fb = load_fragB(bR[j], k0, half);
            c[0][j] = wmma_bf16(a0, fb, c[0][j]);
            c[1][j] = wmma_bf16(a1, fb, c[1][j]);
        }
    }
}

// ---------- prep kernels ----------
__global__ void zero4_kernel(float4* __restrict__ p, int n4) {
    float4 z = make_float4(0.f, 0.f, 0.f, 0.f);
    for (int i = blockIdx.x * blockDim.x + threadIdx.x; i < n4; i += gridDim.x * blockDim.x)
        p[i] = z;
}

__global__ void cvt4_kernel(const float4* __restrict__ src, ushort4* __restrict__ dst, int n4) {
    for (int i = blockIdx.x * blockDim.x + threadIdx.x; i < n4; i += gridDim.x * blockDim.x) {
        float4 f = src[i];
        ushort4 o;
        o.x = f2bf(f.x); o.y = f2bf(f.y); o.z = f2bf(f.z); o.w = f2bf(f.w);
        dst[i] = o;
    }
}

// ---------- routing: one wave per token ----------
__global__ void route_score_kernel(const float* __restrict__ x,
                                   const float* __restrict__ gate_w,
                                   const float* __restrict__ gate_b,
                                   const float* __restrict__ bias,
                                   int* __restrict__ sel,      // [T,2]
                                   float* __restrict__ gtmp) { // [T,2]
    int wave = threadIdx.x >> 5;
    int lane = threadIdx.x & 31;
    int t = blockIdx.x * (blockDim.x >> 5) + wave;
    if (t >= T) return;

    float acc[E];
#pragma unroll
    for (int e = 0; e < E; ++e) acc[e] = 0.f;

    const float* xr = x + (size_t)t * D;
    for (int k = lane; k < D; k += 32) {
        float xv = xr[k];
#pragma unroll
        for (int e = 0; e < E; ++e) acc[e] += xv * gate_w[e * D + k];
    }
#pragma unroll
    for (int e = 0; e < E; ++e)
        for (int off = 16; off > 0; off >>= 1)
            acc[e] += __shfl_xor(acc[e], off, 32);

    if (lane == 0) {
        float s[E];
#pragma unroll
        for (int e = 0; e < E; ++e) s[e] = acc[e] + gate_b[e] + bias[e];
        int e0 = 0;
#pragma unroll
        for (int e = 1; e < E; ++e) if (s[e] > s[e0]) e0 = e;   // first-index tiebreak
        int e1 = -1;
#pragma unroll
        for (int e = 0; e < E; ++e) if (e != e0 && (e1 < 0 || s[e] > s[e1])) e1 = e;
        float x1 = __expf(s[e1] - s[e0]);
        float inv = 1.f / (1.f + x1);
        sel[2 * t]     = e0;  gtmp[2 * t]     = inv;
        sel[2 * t + 1] = e1;  gtmp[2 * t + 1] = x1 * inv;
    }
}

// ---------- deterministic per-expert compaction: one wave per expert ----------
__global__ void compact_kernel(const int* __restrict__ sel, const float* __restrict__ gtmp,
                               int* __restrict__ counts,
                               int* __restrict__ tokList, float* __restrict__ gateList) {
    int e = blockIdx.x;
    int lane = threadIdx.x;
    int pos = 0;
    for (int t0 = 0; t0 < T; t0 += 32) {
        int t = t0 + lane;
        int s0 = sel[2 * t], s1 = sel[2 * t + 1];
        bool m0 = (s0 == e), m1 = (s1 == e);
        bool m = m0 || m1;
        float g = m0 ? gtmp[2 * t] : gtmp[2 * t + 1];
        unsigned mask = (unsigned)__ballot(m);
        if (m) {
            int prefix = __popc(mask & ((1u << lane) - 1u));
            tokList[e * T + pos + prefix]  = t;
            gateList[e * T + pos + prefix] = g;
        }
        pos += __popc(mask);
    }
    if (lane == 0) counts[e] = pos;
}

// ---------- main expert FFN: 32 tokens x one expert per block ----------
// Register tiling 2Mx4N per wave (8 WMMAs / 6KB fragments = 21 FLOP/B), with a
// 2Mx2N tail in stage 1 so each wave covers exactly 352 of H=1408 columns.
__global__ __launch_bounds__(128)
void moe_kernel(const unsigned short* __restrict__ xb,   // [T,D] bf16
                const unsigned short* __restrict__ w1b,  // [E,H,D] bf16
                const unsigned short* __restrict__ w2b,  // [E,D,H] bf16
                const float* __restrict__ b1,            // [E,H]
                const float* __restrict__ b2,            // [E,D]
                const int* __restrict__ counts,
                const int* __restrict__ tokList,
                const float* __restrict__ gateList,
                float* __restrict__ out) {               // [T,D]
    int e = blockIdx.y;
    int cnt = counts[e];
    int base = blockIdx.x * MT;
    if (base >= cnt) return;                 // uniform per block: EXEC stays all-ones

    __shared__ int   sTok[MT];
    __shared__ float sGate[MT];
    __shared__ unsigned short sh[MT * HPAD]; // h tile bf16 (90.6 KB; <=320KB/WG on CDNA5)

    if (threadIdx.x < MT) {
        int idx = base + threadIdx.x;
        if (idx < cnt) {
            sTok[threadIdx.x]  = tokList[e * T + idx];
            sGate[threadIdx.x] = gateList[e * T + idx];
        } else {
            sTok[threadIdx.x]  = 0;      // padded row: gate 0 -> contributes nothing
            sGate[threadIdx.x] = 0.f;
        }
    }
    __syncthreads();

    int wave = threadIdx.x >> 5;
    int lane = threadIdx.x & 31;
    int m    = lane & 15;
    int half = lane >> 4;

    const unsigned short* aRow0 = xb + (size_t)sTok[m]      * D;
    const unsigned short* aRow1 = xb + (size_t)sTok[m + 16] * D;

    // ---- Stage 1: h[32,H] = silu(X . w1[e]^T + b1[e]) ----
    // 5 quad-tiles (64 cols) per wave over cols [0,1280)
    for (int nq = wave; nq < 20; nq += NWAVES) {
        int n0 = nq * 64;
        v8f c[2][4];
#pragma unroll
        for (int i = 0; i < 2; ++i)
#pragma unroll
            for (int j = 0; j < 4; ++j) c[i][j] = (v8f){};
        gemm2xN<4>(aRow0, aRow1, w1b + ((size_t)e * H + n0 + m) * D,
                   (size_t)16 * D, D, half, c);
        float bv[4];
#pragma unroll
        for (int j = 0; j < 4; ++j) bv[j] = b1[e * H + n0 + j * 16 + m];
#pragma unroll
        for (int r = 0; r < 8; ++r) {
            int M0 = r + half * 8, M1 = M0 + 16;
#pragma unroll
            for (int j = 0; j < 4; ++j) {
                float s0 = c[0][j][r] + bv[j];
                float s1 = c[1][j][r] + bv[j];
                sh[M0 * HPAD + n0 + j * 16 + m] =
                    f2bf(s0 * __builtin_amdgcn_rcpf(1.f + __expf(-s0)));
                sh[M1 * HPAD + n0 + j * 16 + m] =
                    f2bf(s1 * __builtin_amdgcn_rcpf(1.f + __expf(-s1)));
            }
        }
    }
    // tail: one pair-tile (32 cols) per wave over cols [1280,1408)
    {
        int n0 = 1280 + wave * 32;
        v8f c[2][2];
#pragma unroll
        for (int i = 0; i < 2; ++i)
#pragma unroll
            for (int j = 0; j < 2; ++j) c[i][j] = (v8f){};
        gemm2xN<2>(aRow0, aRow1, w1b + ((size_t)e * H + n0 + m) * D,
                   (size_t)16 * D, D, half, c);
        float bv[2];
#pragma unroll
        for (int j = 0; j < 2; ++j) bv[j] = b1[e * H + n0 + j * 16 + m];
#pragma unroll
        for (int r = 0; r < 8; ++r) {
            int M0 = r + half * 8, M1 = M0 + 16;
#pragma unroll
            for (int j = 0; j < 2; ++j) {
                float s0 = c[0][j][r] + bv[j];
                float s1 = c[1][j][r] + bv[j];
                sh[M0 * HPAD + n0 + j * 16 + m] =
                    f2bf(s0 * __builtin_amdgcn_rcpf(1.f + __expf(-s0)));
                sh[M1 * HPAD + n0 + j * 16 + m] =
                    f2bf(s1 * __builtin_amdgcn_rcpf(1.f + __expf(-s1)));
            }
        }
    }
    __syncthreads();

    // ---- Stage 2: y[32,D] = h . w2[e]^T + b2[e]; out[tok] += gate * y ----
    const unsigned short* aL0 = sh + m * HPAD;          // LDS rows, M-subtile 0
    const unsigned short* aL1 = sh + (m + 16) * HPAD;   // LDS rows, M-subtile 1
    for (int nq = wave; nq < 32; nq += NWAVES) {        // 8 quad-tiles per wave
        int n0 = nq * 64;
        v8f c[2][4];
#pragma unroll
        for (int i = 0; i < 2; ++i)
#pragma unroll
            for (int j = 0; j < 4; ++j) c[i][j] = (v8f){};
        gemm2xN<4>(aL0, aL1, w2b + ((size_t)e * D + n0 + m) * H,
                   (size_t)16 * H, H, half, c);
        float bv[4];
#pragma unroll
        for (int j = 0; j < 4; ++j) bv[j] = b2[e * D + n0 + j * 16 + m];
#pragma unroll
        for (int r = 0; r < 8; ++r) {
            int M0 = r + half * 8, M1 = M0 + 16;
            float g0 = sGate[M0], g1 = sGate[M1];
            size_t o0 = (size_t)sTok[M0] * D + n0 + m;
            size_t o1 = (size_t)sTok[M1] * D + n0 + m;
#pragma unroll
            for (int j = 0; j < 4; ++j) {
                // exactly 2 commutative fp32 adds per out element -> deterministic
                unsafeAtomicAdd(&out[o0 + j * 16], (c[0][j][r] + bv[j]) * g0);
                unsafeAtomicAdd(&out[o1 + j * 16], (c[1][j][r] + bv[j]) * g1);
            }
        }
    }
}

// ---------- host launcher ----------
extern "C" void kernel_launch(void* const* d_in, const int* in_sizes, int n_in,
                              void* d_out, int out_size, void* d_ws, size_t ws_size,
                              hipStream_t stream) {
    const float* x      = (const float*)d_in[0];
    const float* gate_w = (const float*)d_in[1];
    const float* gate_b = (const float*)d_in[2];
    const float* bias   = (const float*)d_in[3];
    const float* w1     = (const float*)d_in[4];
    const float* b1     = (const float*)d_in[5];
    const float* w2     = (const float*)d_in[6];
    const float* b2     = (const float*)d_in[7];
    float* out = (float*)d_out;

    char* ws = (char*)d_ws;
    // 256B-aligned workspace layout
    int*            counts   = (int*)           (ws + 0);          //   32 B
    int*            sel      = (int*)           (ws + 256);        //  32 KB  [T,2]
    float*          gtmp     = (float*)         (ws + 33024);      //  32 KB  [T,2]
    int*            tokList  = (int*)           (ws + 65792);      // 128 KB  [E,T]
    float*          gateList = (float*)         (ws + 196864);     // 128 KB  [E,T]
    unsigned short* xb       = (unsigned short*)(ws + 327936);     //  16 MB  [T,D]
    unsigned short* w1b      = (unsigned short*)(ws + 17105152);   //  44 MB  [E,H,D]
    unsigned short* w2b      = (unsigned short*)(ws + 63242496);   //  44 MB  [E,D,H]

    // 1) zero output (harness poisons it)
    zero4_kernel<<<2048, 256, 0, stream>>>((float4*)out, (T * D) / 4);
    // 2) fp32 -> bf16 conversions
    cvt4_kernel<<<2048, 256, 0, stream>>>((const float4*)x,  (ushort4*)xb,  (T * D) / 4);
    cvt4_kernel<<<4096, 256, 0, stream>>>((const float4*)w1, (ushort4*)w1b, (E * H * D) / 4);
    cvt4_kernel<<<4096, 256, 0, stream>>>((const float4*)w2, (ushort4*)w2b, (E * D * H) / 4);
    // 3) gating + top-2 softmax (one wave per token)
    route_score_kernel<<<T / 4, 128, 0, stream>>>(x, gate_w, gate_b, bias, sel, gtmp);
    // 4) deterministic per-expert compaction
    compact_kernel<<<E, 32, 0, stream>>>(sel, gtmp, counts, tokList, gateList);
    // 5) expert FFN with WMMA bf16; grid covers worst-case 128 tiles per expert
    moe_kernel<<<dim3(T / MT, E), 128, 0, stream>>>(xb, w1b, w2b, b1, b2,
                                                    counts, tokList, gateList, out);
}
